// MultiHeadAttention_2946347565884
// MI455X (gfx1250) — matrix operand order repaired and checked by reading, not compiled, not run on hardware
//
#include <hip/hip_runtime.h>
#include <stddef.h>
#include <stdint.h>

// ---------------------------------------------------------------------------
// MHA forward for MI455X (gfx1250): bf16 WMMA everywhere, two-pass softmax,
// probs-mean accumulated in LDS, async-tensor V staging, context in bf16 ws.
// ---------------------------------------------------------------------------

#define D_MODEL 1024
#define NUM_HEADS 16
#define D_K 64
#define SEQ 2048
#define BATCH 4
#define SCALE 0.125f
#define NEG_INF -1.0e9f

typedef __bf16 bf16_t;
typedef bf16_t v16bf __attribute__((ext_vector_type(16)));
typedef float  v8f   __attribute__((ext_vector_type(8)));
typedef __attribute__((address_space(3))) unsigned short lds_u16;

union AB  { v16bf v; unsigned short s[16]; };
union ACC { v8f   v; float          f[8];  };

__device__ __forceinline__ unsigned short f2bf(float f) {
    unsigned int u = __float_as_uint(f);
    u += 0x7fffu + ((u >> 16) & 1u);        // round-to-nearest-even
    return (unsigned short)(u >> 16);
}

__device__ __forceinline__ v8f wmma_bf16(v16bf a, v16bf b, v8f c) {
    return __builtin_amdgcn_wmma_f32_16x16x32_bf16(false, a, false, b,
                                                   (short)0, c, false, false);
}

// xor-shuffle within 16-lane rows: prefer v_permlane16 (VALU, co-executes
// with XDL WMMA) over ds_bpermute (DS pipe, contended by atomics/staging).
__device__ __forceinline__ float lane_xor16(float v, int o,
                                            unsigned L, unsigned H) {
#if __has_builtin(__builtin_amdgcn_permlane16)
    unsigned u = __float_as_uint(v);
    return __uint_as_float(
        (unsigned)__builtin_amdgcn_permlane16(u, u, L, H, false, false));
#else
    return __shfl_xor(v, o, 16);
#endif
}
__device__ __forceinline__ float red16_max(float v) {
    v = fmaxf(v, lane_xor16(v, 1, 0x67452301u, 0xEFCDAB89u));
    v = fmaxf(v, lane_xor16(v, 2, 0x54761032u, 0xDCFE98BAu));
    v = fmaxf(v, lane_xor16(v, 4, 0x32107654u, 0xBA98FEDCu));
    v = fmaxf(v, lane_xor16(v, 8, 0xFEDCBA98u, 0x76543210u));
    return v;
}
__device__ __forceinline__ float red16_sum(float v) {
    v += lane_xor16(v, 1, 0x67452301u, 0xEFCDAB89u);
    v += lane_xor16(v, 2, 0x54761032u, 0xDCFE98BAu);
    v += lane_xor16(v, 4, 0x32107654u, 0xBA98FEDCu);
    v += lane_xor16(v, 8, 0xFEDCBA98u, 0x76543210u);
    return v;
}

// ---------------------------------------------------------------------------
// GEMM: C[M,N] = A[M,K] * B[N,K]^T  (torch Linear semantics), K contraction.
// Tile 128x64, 8 waves of 32x32, K-step 32, bf16 conversion in LDS staging.
// ---------------------------------------------------------------------------
template<bool A_BF16, bool OUT_F32>
__global__ __launch_bounds__(256)
void gemm_tn(const void* __restrict__ Ain, const float* __restrict__ Bw,
             void* __restrict__ Cout, const float* __restrict__ bias,
             int M, int N, int K)
{
    __shared__ unsigned short As[128][40];   // padded bf16 tile
    __shared__ unsigned short Bs[64][40];

    const int tid  = threadIdx.x;
    const int m0   = blockIdx.y * 128;
    const int n0   = blockIdx.x * 64;
    const int w    = tid >> 5;
    const int lane = tid & 31;
    const int lr   = lane & 15;
    const int hi   = lane >> 4;
    const int mbase = (w >> 1) * 32;
    const int nbase = (w & 1) * 32;

    const float*          Af = (const float*)Ain;
    const unsigned short* Au = (const unsigned short*)Ain;

    ACC acc[2][2];
#pragma unroll
    for (int i = 0; i < 2; ++i)
#pragma unroll
        for (int j = 0; j < 2; ++j)
#pragma unroll
            for (int r = 0; r < 8; ++r) acc[i][j].f[r] = 0.0f;

    for (int k0 = 0; k0 < K; k0 += 32) {
#pragma unroll
        for (int i = 0; i < 16; ++i) {
            int idx = tid + i * 256;
            int row = idx >> 5, col = idx & 31;
            size_t g = (size_t)(m0 + row) * K + (k0 + col);
            As[row][col] = A_BF16 ? Au[g] : f2bf(Af[g]);
        }
#pragma unroll
        for (int i = 0; i < 8; ++i) {
            int idx = tid + i * 256;
            int row = idx >> 5, col = idx & 31;
            Bs[row][col] = f2bf(Bw[(size_t)(n0 + row) * K + (k0 + col)]);
        }
        if (k0 + 32 < K) {
            const char* base = (const char*)Ain;
            size_t elt = A_BF16 ? 2 : 4;
            __builtin_prefetch(base + ((size_t)(m0 + (tid & 127)) * K + k0 + 32) * elt, 0, 1);
        }
        __syncthreads();

        AB a[2], bfr[2];
#pragma unroll
        for (int ms = 0; ms < 2; ++ms) {
            int row = mbase + ms * 16 + lr;
#pragma unroll
            for (int j = 0; j < 8; ++j) {
                a[ms].s[j]     = As[row][hi * 8 + j];
                a[ms].s[8 + j] = As[row][16 + hi * 8 + j];
            }
        }
#pragma unroll
        for (int ns = 0; ns < 2; ++ns) {
            int row = nbase + ns * 16 + lr;
#pragma unroll
            for (int j = 0; j < 8; ++j) {
                bfr[ns].s[j]     = Bs[row][hi * 8 + j];
                bfr[ns].s[8 + j] = Bs[row][16 + hi * 8 + j];
            }
        }
#pragma unroll
        for (int ms = 0; ms < 2; ++ms)
#pragma unroll
            for (int ns = 0; ns < 2; ++ns)
                acc[ms][ns].v = wmma_bf16(a[ms].v, bfr[ns].v, acc[ms][ns].v);
        __syncthreads();
    }

#pragma unroll
    for (int ms = 0; ms < 2; ++ms)
#pragma unroll
        for (int ns = 0; ns < 2; ++ns) {
            int coln = n0 + nbase + ns * 16 + lr;
            float bv = 0.0f;
            if (OUT_F32) bv = bias[coln];
#pragma unroll
            for (int r = 0; r < 8; ++r) {
                int rowm = m0 + mbase + ms * 16 + r + (hi << 3);
                size_t g = (size_t)rowm * N + coln;
                if (OUT_F32) ((float*)Cout)[g] = acc[ms][ns].f[r] + bv;
                else ((unsigned short*)Cout)[g] = f2bf(acc[ms][ns].f[r]);
            }
        }
}

// ---------------------------------------------------------------------------
// Attention
// ---------------------------------------------------------------------------
__device__ __forceinline__
void score_frags(const unsigned short* __restrict__ Kb,
                 const int* __restrict__ mask, const AB* aq,
                 int b, int h, int kbase, int lr, int hi,
                 float* f0, float* f1)
{
#pragma unroll
    for (int ns = 0; ns < 2; ++ns) {
        const int col = kbase + ns * 16 + lr;
        ACC acc;
#pragma unroll
        for (int r = 0; r < 8; ++r) acc.f[r] = 0.0f;
#pragma unroll
        for (int dstep = 0; dstep < 2; ++dstep) {
            AB bk;
            const unsigned short* kp = Kb + ((size_t)(b * SEQ + col)) * D_MODEL
                                          + h * D_K + dstep * 32;
#pragma unroll
            for (int j = 0; j < 8; ++j) {
                bk.s[j]     = kp[hi * 8 + j];
                bk.s[8 + j] = kp[16 + hi * 8 + j];
            }
            acc.v = wmma_bf16(aq[dstep].v, bk.v, acc.v);
        }
        const bool keep = (mask[b * SEQ + col] != 0);
        float* f = ns ? f1 : f0;
#pragma unroll
        for (int r = 0; r < 8; ++r)
            f[r] = keep ? acc.f[r] * SCALE : NEG_INF;
    }
}

__global__ __launch_bounds__(128)
void attn_kernel(const unsigned short* __restrict__ Qb,
                 const unsigned short* __restrict__ Kb,
                 const unsigned short* __restrict__ Vb,
                 const int* __restrict__ mask,
                 unsigned short* __restrict__ Ctx,
                 float* __restrict__ probs_out)
{
    __shared__ float pm[16][SEQ];                          // probs-mean (128 KB)
    __shared__ unsigned short stage[4][16][32];            // P transpose slabs
    __shared__ __align__(16) unsigned short vslab[4][32][72]; // V tiles, padded

    const int b     = blockIdx.x;
    const int qbase = blockIdx.y * 16;
    const int tid   = threadIdx.x;
    const int w     = tid >> 5;
    const int lane  = tid & 31;
    const int lr    = lane & 15;
    const int hi    = lane >> 4;

    // per-wave LDS byte offset of the V slab (for async-to-LDS addressing)
    const unsigned vbase =
        (unsigned)(uintptr_t)(lds_u16*)&vslab[w][0][0];
    const int vrow_in  = lane >> 3;          // 0..3
    const int vcolb    = (lane & 7) * 16;    // byte offset within 128B row

    // zero the probs-mean accumulator
    float4* pm4 = reinterpret_cast<float4*>(&pm[0][0]);
#pragma unroll 4
    for (int i = 0; i < 64; ++i)
        pm4[tid + i * 128] = make_float4(0.f, 0.f, 0.f, 0.f);
    __syncthreads();

    for (int hh = 0; hh < 4; ++hh) {
        const int h = hh * 4 + w;

        // Q frags (reused over all 2048 keys)
        AB aq[2];
#pragma unroll
        for (int dstep = 0; dstep < 2; ++dstep) {
            const unsigned short* qp = Qb + ((size_t)(b * SEQ + qbase + lr)) * D_MODEL
                                          + h * D_K + dstep * 32;
#pragma unroll
            for (int j = 0; j < 8; ++j) {
                aq[dstep].s[j]     = qp[hi * 8 + j];
                aq[dstep].s[8 + j] = qp[16 + hi * 8 + j];
            }
        }

        float m[8], l[8];
#pragma unroll
        for (int r = 0; r < 8; ++r) { m[r] = -3.0e38f; l[r] = 0.0f; }

        // ---- pass 1: softmax statistics ----
        for (int kt = 0; kt < SEQ / 32; ++kt) {
            float f0[8], f1[8];
            score_frags(Kb, mask, aq, b, h, kt * 32, lr, hi, f0, f1);
#pragma unroll
            for (int r = 0; r < 8; ++r) {
                float v = red16_max(fmaxf(f0[r], f1[r]));
                float nm = fmaxf(m[r], v);
                float e = red16_sum(__expf(f0[r] - nm) + __expf(f1[r] - nm));
                l[r] = l[r] * __expf(m[r] - nm) + e;
                m[r] = nm;
            }
        }
        float linv[8];
#pragma unroll
        for (int r = 0; r < 8; ++r) linv[r] = 1.0f / l[r];

        // ---- pass 2: probs, probs-mean, P*V ----
        ACC ctx[4];
#pragma unroll
        for (int n = 0; n < 4; ++n)
#pragma unroll
            for (int r = 0; r < 8; ++r) ctx[n].f[r] = 0.0f;

        for (int kt = 0; kt < SEQ / 32; ++kt) {
            const int kbase = kt * 32;

            // --- async-stage this k-tile's V block (32 keys x 64 dims bf16)
            // into the wave-private LDS slab; overlaps the score WMMAs below.
            asm volatile("s_wait_dscnt 0" ::: "memory");   // WAR on vslab reads
            {
                const unsigned short* vrow = Vb
                    + ((size_t)(b * SEQ + kbase)) * D_MODEL + h * D_K;
#pragma unroll
                for (int i = 0; i < 8; ++i) {
                    int row = i * 4 + vrow_in;
                    unsigned laddr = vbase + (unsigned)(row * 144 + vcolb);
                    unsigned long long gaddr = (unsigned long long)(uintptr_t)vrow
                        + (unsigned long long)row * (D_MODEL * 2) + vcolb;
                    asm volatile("global_load_async_to_lds_b128 %0, %1, off"
                                 :: "v"(laddr), "v"(gaddr) : "memory");
                }
            }

            float f0[8], f1[8];
            score_frags(Kb, mask, aq, b, h, kbase, lr, hi, f0, f1);
#pragma unroll
            for (int r = 0; r < 8; ++r) {
                float p0 = __expf(f0[r] - m[r]) * linv[r];
                float p1 = __expf(f1[r] - m[r]) * linv[r];
                int mr = r + (hi << 3);
                atomicAdd(&pm[mr][kbase + lr],      p0 * (1.0f / NUM_HEADS));
                atomicAdd(&pm[mr][kbase + 16 + lr], p1 * (1.0f / NUM_HEADS));
                stage[w][mr][lr]      = f2bf(p0);   // D-layout -> row-major
                stage[w][mr][16 + lr] = f2bf(p1);
            }
            asm volatile("s_wait_dscnt 0" ::: "memory");  // stage slab RAW

            // reload P as A-frag (row = lr, K chunks hi*8 / 16+hi*8)
            AB ap;
#pragma unroll
            for (int j = 0; j < 8; ++j) {
                ap.s[j]     = stage[w][lr][hi * 8 + j];
                ap.s[8 + j] = stage[w][lr][16 + hi * 8 + j];
            }

            // V B-frags from the async-staged LDS slab
            asm volatile("s_wait_asynccnt 0" ::: "memory");
#pragma unroll
            for (int n = 0; n < 4; ++n) {
                AB bv;
#pragma unroll
                for (int j = 0; j < 8; ++j) {
                    bv.s[j]     = vslab[w][hi * 8 + j][n * 16 + lr];
                    bv.s[8 + j] = vslab[w][16 + hi * 8 + j][n * 16 + lr];
                }
                ctx[n].v = wmma_bf16(ap.v, bv.v, ctx[n].v);
            }
        }

        // store context tile (bf16) at [b, q, h*64 + d]
#pragma unroll
        for (int n = 0; n < 4; ++n)
#pragma unroll
            for (int r = 0; r < 8; ++r) {
                int mr = r + (hi << 3);
                Ctx[((size_t)(b * SEQ + qbase + mr)) * D_MODEL + h * D_K + n * 16 + lr]
                    = f2bf(ctx[n].f[r]);
            }
    }
    __syncthreads();

    // probs-mean tile is contiguous in global: rows qbase..qbase+15
    const float4* pmr = reinterpret_cast<const float4*>(&pm[0][0]);
    float4* out4 = reinterpret_cast<float4*>(probs_out + ((size_t)b * SEQ + qbase) * SEQ);
#pragma unroll 4
    for (int i = 0; i < 64; ++i)
        out4[tid + i * 128] = pmr[tid + i * 128];
}

// ---------------------------------------------------------------------------
// launch
// ---------------------------------------------------------------------------
extern "C" void kernel_launch(void* const* d_in, const int* in_sizes, int n_in,
                              void* d_out, int out_size, void* d_ws, size_t ws_size,
                              hipStream_t stream)
{
    const float* query = (const float*)d_in[0];
    const float* key   = (const float*)d_in[1];
    const float* value = (const float*)d_in[2];
    const int*   mask  = (const int*)d_in[3];
    const float* Wq    = (const float*)d_in[4];
    const float* Wk    = (const float*)d_in[5];
    const float* Wv    = (const float*)d_in[6];
    const float* Wo    = (const float*)d_in[7];
    const float* bo    = (const float*)d_in[8];

    float* out   = (float*)d_out;
    float* probs = out + (size_t)BATCH * SEQ * D_MODEL;

    const size_t SZ = (size_t)BATCH * SEQ * D_MODEL;   // elements per bf16 buffer
    unsigned short* Qb = (unsigned short*)d_ws;
    unsigned short* Kb = Qb + SZ;
    unsigned short* Vb = Kb + SZ;
    unsigned short* Cb = Vb + SZ;

    const int M = BATCH * SEQ;          // 8192
    const int N = D_MODEL, K = D_MODEL; // 1024
    dim3 gemm_grid(N / 64, M / 128);    // 16 x 64
    dim3 gemm_block(256);

    gemm_tn<false, false><<<gemm_grid, gemm_block, 0, stream>>>(query, Wq, Qb, nullptr, M, N, K);
    gemm_tn<false, false><<<gemm_grid, gemm_block, 0, stream>>>(key,   Wk, Kb, nullptr, M, N, K);
    gemm_tn<false, false><<<gemm_grid, gemm_block, 0, stream>>>(value, Wv, Vb, nullptr, M, N, K);

    attn_kernel<<<dim3(BATCH, SEQ / 16), 128, 0, stream>>>(Qb, Kb, Vb, mask, Cb, probs);

    gemm_tn<true, true><<<gemm_grid, gemm_block, 0, stream>>>(Cb, Wo, out, bo, M, N, K);
}
